// Maj3_43885975830689
// MI455X (gfx1250) — 8-bit matrix operand order repaired and checked by reading, compiled, not run on hardware
//
#include <hip/hip_runtime.h>
#include <stdint.h>

typedef __attribute__((ext_vector_type(8))) int v8i;

#define CIN   128
#define COUT  128
#define HH    16
#define WW    16
#define BB    8
#define KTOT  1536
#define M_TOT 2048
#define NKSTEP (KTOT / 64)

__device__ __forceinline__ int8_t sgn8(float v) {
    return (int8_t)((v > 0.f) ? 1 : ((v < 0.f) ? -1 : 0));
}

__global__ void maj3_pack_w(const float* __restrict__ w, int8_t* __restrict__ Bp) {
    int idx = blockIdx.x * blockDim.x + threadIdx.x;
    if (idx >= COUT * 3 * CIN) return;
    int c  = idx & (CIN - 1);
    int ky = (idx >> 7) % 3;
    int o  = idx / (3 * CIN);
    int8_t s0 = sgn8(w[((o * 3 + ky) * 3 + 0) * CIN + c]);
    int8_t s1 = sgn8(w[((o * 3 + ky) * 3 + 1) * CIN + c]);
    int8_t s2 = sgn8(w[((o * 3 + ky) * 3 + 2) * CIN + c]);
    int8_t* row = Bp + (size_t)o * KTOT;
    row[(ky * 3 + 0) * CIN + c] = s0;
    row[(ky * 3 + 1) * CIN + c] = s1;
    row[(ky * 3 + 2) * CIN + c] = s2;
    row[9 * CIN + ky * CIN + c] = (int8_t)(-(s0 * s1 * s2));
}

__global__ void maj3_pack_x(const float* __restrict__ x, int8_t* __restrict__ Ap) {
    int idx = blockIdx.x * blockDim.x + threadIdx.x;
    if (idx >= M_TOT * CIN) return;
    int c = idx & (CIN - 1);
    int p = idx >> 7;
    int b  = p >> 8;
    int h  = (p >> 4) & 15;
    int wc = p & 15;
    int8_t* row = Ap + (size_t)p * KTOT;
    const float* xbc = x + ((size_t)(b * CIN + c)) * (HH * WW);
    #pragma unroll
    for (int ky = 0; ky < 3; ++ky) {
        int y = h + ky - 1;
        int8_t s[3];
        if (y >= 0 && y < HH) {
            #pragma unroll
            for (int kx = 0; kx < 3; ++kx) {
                int xc = wc + kx - 1;
                s[kx] = (xc >= 0 && xc < WW) ? sgn8(xbc[y * WW + xc]) : (int8_t)0;
            }
        } else {
            s[0] = s[1] = s[2] = 0;
        }
        row[(ky * 3 + 0) * CIN + c] = s[0];
        row[(ky * 3 + 1) * CIN + c] = s[1];
        row[(ky * 3 + 2) * CIN + c] = s[2];
        row[9 * CIN + ky * CIN + c] = (int8_t)(s[0] * s[1] * s[2]);
    }
}

__global__ __launch_bounds__(256) void maj3_gemm(const int8_t* __restrict__ Ap,
                                                 const int8_t* __restrict__ Bp,
                                                 float* __restrict__ out) {
    int lane = threadIdx.x & 31;
    int wave = (blockIdx.x * blockDim.x + threadIdx.x) >> 5;
    int m_base = (wave >> 3) * 16;
    int n_base = (wave & 7) * 16;
    int lo16 = lane & 15;
    int hi   = lane >> 4;

    const uint8_t* arow = (const uint8_t*)Ap + (size_t)(m_base + lo16) * KTOT + hi * 8;
    const uint8_t* bcol = (const uint8_t*)Bp + (size_t)(n_base + lo16) * KTOT + hi * 8;

    v8i acc = {};
    #pragma unroll 4
    for (int ks = 0; ks < NKSTEP; ++ks) {
        const uint32_t* ap = (const uint32_t*)(arow + ks * 64);
        const uint32_t* bp = (const uint32_t*)(bcol + ks * 64);
        v8i A, B;
        A[0] = ap[0];  A[1] = ap[1];
        A[2] = ap[4];  A[3] = ap[5];
        A[4] = ap[8];  A[5] = ap[9];
        A[6] = ap[12]; A[7] = ap[13];
        B[0] = bp[0];  B[1] = bp[1];  B[2] = bp[4];  B[3] = bp[5];
        B[4] = bp[8];  B[5] = bp[9];  B[6] = bp[12]; B[7] = bp[13];
        acc = __builtin_amdgcn_wmma_i32_16x16x64_iu8(true, A, true, B, acc,
                                                     false, false);
    }

    int o = n_base + lo16;
    #pragma unroll
    for (int v = 0; v < 8; ++v) {
        int m = v + hi * 8;
        int p = m_base + m;
        int b  = p >> 8;
        int h  = (p >> 4) & 15;
        int wc = p & 15;
        out[(((size_t)b * COUT + o) * HH + h) * WW + wc] = 0.5f * (float)acc[v];
    }
}

extern "C" void kernel_launch(void* const* d_in, const int* in_sizes, int n_in,
                              void* d_out, int out_size, void* d_ws, size_t ws_size,
                              hipStream_t stream) {
    const float* x   = (const float*)d_in[0];
    const float* wgt = (const float*)d_in[1];
    float* out = (float*)d_out;

    int8_t* Ap = (int8_t*)d_ws;
    int8_t* Bp = Ap + (size_t)M_TOT * KTOT;

    maj3_pack_w<<<(COUT * 3 * CIN + 255) / 256, 256, 0, stream>>>(wgt, Bp);
    maj3_pack_x<<<(M_TOT * CIN + 255) / 256, 256, 0, stream>>>(x, Ap);
    maj3_gemm<<<128, 256, 0, stream>>>(Ap, Bp, out);
}
